// VectorQuantizer_9156870275434
// MI455X (gfx1250) — matrix operand order repaired and checked by reading, not compiled, not run on hardware
//
#include <hip/hip_runtime.h>

typedef __bf16 bf16;
typedef __attribute__((ext_vector_type(16))) __bf16 v16bf;
typedef __attribute__((ext_vector_type(8)))  float  v8f;
typedef __attribute__((ext_vector_type(4)))  float  v4f;

#define NUM_EMB   1024
#define EMB_DIM   256
#define SPAT      1024        // 32*32 spatial positions per batch image
#define TOTAL_OUT 8388608     // 32*256*32*32

// workspace layout (bytes)
#define EMBB_OFF  0           // 64 tiles * 8 ksteps * 32 lanes * 16 bf16 = 524288 B
#define NRM_OFF   524288      // 1024 floats
#define BSUM_OFF  528384      // 256 floats

// LDS staging geometry for B tiles: per k-step group: 32 lanes * 32B + swizzle pad
#define SGRP      1120        // 32*32 + 3*32 pad (32B-aligned swizzle)
#define STAGE_SZ  (8 * SGRP)  // 8960 B per stage
#define NSTAGE_SZ (2 * STAGE_SZ)

// ---------------------------------------------------------------------------
// Prep 1: codebook -> bf16, pre-swizzled into WMMA B-matrix per-lane layout.
// B (32x16 bf16): lanes 0-15 hold column N=lane, K=0..7,16..23; lanes 16-31 hold
// column N=lane-16, K=8..15,24..31. Each lane's 16 bf16 stored contiguously.
// ---------------------------------------------------------------------------
__global__ __launch_bounds__(256) void vq_prep_embB(const float* __restrict__ emb,
                                                    bf16* __restrict__ embB) {
  int i = blockIdx.x * 256 + threadIdx.x;          // 0 .. 262143
  int e = i & 15;
  int l = (i >> 4) & 31;
  int s = (i >> 9) & 7;                            // k-step (32 dims each)
  int c = i >> 12;                                 // column tile (16 codes)
  int k = c * 16 + (l & 15);                       // code index
  int d = s * 32 + ((l >> 4) << 3) + ((e < 8) ? e : (e + 8)); // dim index
  embB[i] = (bf16)emb[k * EMB_DIM + d];
}

// Prep 2: ||e_k||^2
__global__ __launch_bounds__(256) void vq_prep_nrm(const float* __restrict__ emb,
                                                   float* __restrict__ nrm) {
  int k = blockIdx.x * 256 + threadIdx.x;
  if (k < NUM_EMB) {
    float s = 0.f;
    for (int d = 0; d < EMB_DIM; ++d) {
      float v = emb[k * EMB_DIM + d];
      s += v * v;
    }
    nrm[k] = s;
  }
}

// Async copy of one 8KB B column tile into an LDS stage.
// Thread t (0..255): s = t/32, l = t%32 copies its lane's 32B chunk.
// IOFFSET applies to BOTH lds and global address, so one base pair serves
// the two b128 halves.
__device__ __forceinline__ void issue_b_tile(const bf16* embB, int c,
                                             unsigned ldsStageBase, int tid) {
  int s = tid >> 5;
  int l = tid & 31;
  unsigned loff = ldsStageBase + (unsigned)(s * SGRP + l * 32 + ((l >> 3) << 5));
  unsigned long long g = (unsigned long long)(size_t)embB +
                         (unsigned long long)(c * 8192 + s * 1024 + l * 32);
  asm volatile("global_load_async_to_lds_b128 %0, %1, off\n\t"
               "global_load_async_to_lds_b128 %0, %1, off offset:16"
               :: "v"(loff), "v"(g) : "memory");
}

// ---------------------------------------------------------------------------
// Main: each wave owns a 16-row M-tile; A fragments live in VGPRs.
// B tiles double-buffered in LDS via async copies (ASYNCcnt).
// ---------------------------------------------------------------------------
__global__ __launch_bounds__(256) void vq_main(const float* __restrict__ z,
                                               const float* __restrict__ emb,
                                               const bf16*  __restrict__ embB,
                                               const float* __restrict__ nrm,
                                               float* __restrict__ out,
                                               float* __restrict__ blockSums) {
  __shared__ __align__(32) char sB[NSTAGE_SZ];     // 2 x 8960 B B-tile stages
  __shared__ int   sIdx[8 * 16];
  __shared__ float sWave[8];

  const int tid  = threadIdx.x;
  const int wave = tid >> 5;
  const int l    = tid & 31;
  const int row  = l & 15;         // M within tile (also N within C tile)
  const int hi   = l >> 4;         // half-lane select
  const int n0   = blockIdx.x * 128 + wave * 16;   // first flat row of tile
  const int bb   = n0 >> 10;                       // batch index
  const int hwb  = n0 & 1023;                      // spatial base (16-aligned)

  const unsigned sbBase = (unsigned)(size_t)&sB[0];   // LDS byte offset

  // kick off tile 0 into stage 0 before touching z (overlap with A-build)
  issue_b_tile(embB, 0, sbBase, tid);

  // ---- build A fragments (16-bit A 16x32 layout) + per-lane fp32 ||z||^2 ----
  v16bf a[8];
  float znormPart = 0.f;
  #pragma unroll
  for (int s = 0; s < 8; ++s) {
    #pragma unroll
    for (int e = 0; e < 16; ++e) {
      int klocal = ((e < 8) ? e : (e + 8)) + (hi << 3);
      int d = s * 32 + klocal;
      float zv = z[(bb * EMB_DIM + d) * SPAT + hwb + row];
      znormPart += zv * zv;
      a[s][e] = (bf16)zv;
    }
  }

  // ---- running argmin over 64 column tiles ----
  float bestD[8];
  int   bestI[8];
  #pragma unroll
  for (int j = 0; j < 8; ++j) { bestD[j] = 3.4e38f; bestI[j] = 0; }

  for (int c = 0; c < 64; ++c) {
    asm volatile("s_wait_asynccnt 0x0" ::: "memory");
    __syncthreads();                                // tile c resident in LDS
    if (c < 63) issue_b_tile(embB, c + 1, sbBase + ((c + 1) & 1) * STAGE_SZ, tid);

    const char* stage = sB + (c & 1) * STAGE_SZ;
    v8f acc = {};
    #pragma unroll
    for (int s = 0; s < 8; ++s) {
      v16bf bv = *(const v16bf*)(stage + s * SGRP + l * 32 + ((l >> 3) << 5));
      acc = __builtin_amdgcn_wmma_f32_16x16x32_bf16(false, a[s], false, bv,
                                                    (short)0, acc, false, false);
    }
    // C layout: lane holds column N = l%16; VGPR j holds row M = j + 8*(l/16)
    int   col = c * 16 + row;
    float nv  = nrm[col];
    #pragma unroll
    for (int j = 0; j < 8; ++j) {
      float dist = nv - 2.0f * acc[j];              // ||e||^2 - 2 z.e
      if (dist < bestD[j]) { bestD[j] = dist; bestI[j] = col; }
    }
  }

  // ---- argmin reduce across the 16 lanes holding one row (ties -> low idx) ----
  #pragma unroll
  for (int j = 0; j < 8; ++j) {
    float dj = bestD[j];
    int   ij = bestI[j];
    #pragma unroll
    for (int off = 1; off < 16; off <<= 1) {
      float od = __shfl_xor(dj, off, 32);
      int   oi = __shfl_xor(ij, off, 32);
      if (od < dj || (od == dj && oi < ij)) { dj = od; ij = oi; }
    }
    bestD[j] = dj;
    bestI[j] = ij;
  }

  // rows 0..7 in lane 0, rows 8..15 in lane 16
  if (l == 0) {
    #pragma unroll
    for (int j = 0; j < 8; ++j) sIdx[wave * 16 + j] = bestI[j];
  }
  if (l == 16) {
    #pragma unroll
    for (int j = 0; j < 8; ++j) sIdx[wave * 16 + 8 + j] = bestI[j];
  }

  // ---- loss via algebra: row loss = ||z_row||^2 + bestD_row (no z re-read) ----
  float bsum = 0.f;
  #pragma unroll
  for (int j = 0; j < 8; ++j) bsum += bestD[j];     // identical across 16-lane group
  float lanePart = znormPart + bsum * (1.0f / 16.0f);
  #pragma unroll
  for (int off = 1; off < 32; off <<= 1) lanePart += __shfl_xor(lanePart, off, 32);

  __syncthreads();                                  // sIdx visible to all lanes
  if (l == 0) sWave[wave] = lanePart;

  // ---- write quantized output: lane covers a contiguous 128-dim half-row ----
  const int code = sIdx[wave * 16 + row];
  const float* qrow = emb + code * EMB_DIM + 128 * hi;
  const int outBase = (bb * EMB_DIM + 128 * hi) * SPAT + hwb + row;
  #pragma unroll 4
  for (int k4 = 0; k4 < 32; ++k4) {
    v4f q = *(const v4f*)(qrow + 4 * k4);           // gather from L2-hot codebook
    #pragma unroll
    for (int j = 0; j < 4; ++j)
      out[outBase + (4 * k4 + j) * SPAT] = q[j];    // coalesced across 16 rows
  }

  __syncthreads();
  if (tid == 0) {
    float bs = 0.f;
    for (int w = 0; w < 8; ++w) bs += sWave[w];
    blockSums[blockIdx.x] = bs;
  }
}

// ---------------------------------------------------------------------------
// Finalize: loss = (1 + BETA) * mean((q - z)^2), BETA = 0.25
// ---------------------------------------------------------------------------
__global__ void vq_loss(const float* __restrict__ blockSums, float* __restrict__ out) {
  if (threadIdx.x == 0 && blockIdx.x == 0) {
    float s = 0.f;
    for (int i = 0; i < 256; ++i) s += blockSums[i];
    out[TOTAL_OUT] = 1.25f * s / (float)TOTAL_OUT;
  }
}

extern "C" void kernel_launch(void* const* d_in, const int* in_sizes, int n_in,
                              void* d_out, int out_size, void* d_ws, size_t ws_size,
                              hipStream_t stream) {
  const float* z   = (const float*)d_in[0];   // [32, 256, 32, 32]
  const float* emb = (const float*)d_in[1];   // [1024, 256]
  float* out = (float*)d_out;                 // [8388608 quantized] + [1 loss]
  char*  ws  = (char*)d_ws;

  bf16*  embB = (bf16*)(ws + EMBB_OFF);
  float* nrm  = (float*)(ws + NRM_OFF);
  float* bsum = (float*)(ws + BSUM_OFF);

  vq_prep_embB<<<1024, 256, 0, stream>>>(emb, embB);
  vq_prep_nrm <<<4,    256, 0, stream>>>(emb, nrm);
  vq_main     <<<256,  256, 0, stream>>>(z, emb, embB, nrm, out, bsum);
  vq_loss     <<<1,    1,   0, stream>>>(bsum, out);
}
